// G2N2Layer_18296560681006
// MI455X (gfx1250) — matrix operand order repaired and checked by reading, not compiled, not run on hardware
//
#include <hip/hip_runtime.h>
#include <hip/hip_bf16.h>

// ---------------------------------------------------------------------------
// G2N2 layer for MI455X (gfx1250, wave32, WMMA 16x16x32 f16 -> f32)
// ---------------------------------------------------------------------------

typedef __attribute__((ext_vector_type(16))) _Float16 v16h;
typedef __attribute__((ext_vector_type(8)))  _Float16 v8h;
typedef __attribute__((ext_vector_type(8)))  float    v8f;

#define N_NODES 1024
#define N_EDGES 32768
#define NE      16
#define NN      32
#define NEOUT   16
#define NNOUT   32
#define TMPW    96          // 3*NE + H5 = 80, padded to 96 (3 K-chunks of 32)

// ---- workspace layout (bytes) ----
#define AX_OFF    ((size_t)0)                       // f16 [16][1024][1024] = 32MB (reused as Adense)
#define AYT_OFF   ((size_t)33554432)                // f16 [16][1024][1024] = 32MB (transposed Ay)
#define TMP_OFF   ((size_t)67108864)                // f16 [E][96]          = 6MB
#define H5_OFF    ((size_t)73400320)                // f32 [1024][32]
#define WM1F_OFF  ((size_t)73531392)                // f16 B-frags of Wm1 (96x128): 24 frags * 32 lanes * 16
#define WM2F_OFF  ((size_t)73555968)                // f16 B-frags of Wm2 (128x16): 4 frags
#define XWF_OFF   ((size_t)73560064)                // f16 B-frags of xw[k]=x@aggW[k]: 16*32*2*32*16
#define PART_OFF  ((size_t)74608640)                // f32 [16][1024][32] per-k partials = 2MB

// ---------------------------------------------------------------------------
// fragment load helpers (this file's contiguous-k 16-bit A / B order, wave32)
// A (16xK chunk at m): lane L: M=L%16; halves 0..15 = K m+(0..15)+(L>=16?16:0)
// B (Kx16 chunk):      lane L: N=L%16; halves 0..15 = K base+(0..15)+(L>=16?16:0)
// ---------------------------------------------------------------------------
__device__ __forceinline__ v16h cat8(v8h lo, v8h hi) {
  return __builtin_shufflevector(lo, hi, 0,1,2,3,4,5,6,7,8,9,10,11,12,13,14,15);
}
__device__ __forceinline__ v16h load_a_frag(const _Float16* rowp, int lane, int m) {
  int hs = (lane >= 16) ? 16 : 0;
  v8h lo = *reinterpret_cast<const v8h*>(rowp + m + hs);
  v8h hi = *reinterpret_cast<const v8h*>(rowp + m + hs + 8);
  return cat8(lo, hi);
}
__device__ __forceinline__ v16h load_b_contig(const _Float16* p) {
  v8h lo = *reinterpret_cast<const v8h*>(p);
  v8h hi = *reinterpret_cast<const v8h*>(p + 8);
  return cat8(lo, hi);
}
__device__ __forceinline__ v8f wmma_f16(v16h a, v16h b, v8f c) {
  return __builtin_amdgcn_wmma_f32_16x16x32_f16(false, a, false, b, (short)0, c, false, false);
}

// ---------------------------------------------------------------------------
// utility kernels
// ---------------------------------------------------------------------------
__global__ void zero_u4_kernel(uint4* __restrict__ p, int n) {
  int i = blockIdx.x * blockDim.x + threadIdx.x;
  if (i < n) p[i] = make_uint4(0u, 0u, 0u, 0u);
}

// h5 = (x @ W5) / 32  -> f32 [1024][32]
__global__ void h5_kernel(const float* __restrict__ x, const float* __restrict__ W5,
                          float* __restrict__ h5) {
  int t = blockIdx.x * blockDim.x + threadIdx.x;      // 32768
  int n = t >> 5, o = t & 31;
  float s = 0.f;
  #pragma unroll
  for (int r = 0; r < NN; ++r) s += x[n * NN + r] * W5[r * 32 + o];
  h5[t] = s * (1.0f / NN);
}

// pre-swizzle Wm1 (80x128, zero-padded to 96) and Wm2 (128x16) into B-fragment order
__global__ void weight_frag_kernel(const float* __restrict__ Wm1, const float* __restrict__ Wm2,
                                   _Float16* __restrict__ wm1f, _Float16* __restrict__ wm2f) {
  int t = blockIdx.x * blockDim.x + threadIdx.x;      // 14336
  if (t < 24 * 512) {                                  // Wm1: frag = nt*3 + kcIdx
    int f = t >> 9, rem = t & 511, lane = rem >> 4, h = rem & 15;
    int nt = f / 3, kc = f % 3;
    int K = kc * 32 + h + ((lane < 16) ? 0 : 16);
    int col = nt * 16 + (lane & 15);
    wm1f[t] = (_Float16)((K < 80) ? Wm1[K * 128 + col] : 0.f);
  } else if (t < 24 * 512 + 4 * 512) {                 // Wm2: frag = kcIdx (K chunks of 32)
    int u = t - 24 * 512;
    int f = u >> 9, rem = u & 511, lane = rem >> 4, h = rem & 15;
    int K = f * 32 + h + ((lane < 16) ? 0 : 16);
    int col = lane & 15;
    wm2f[u] = (_Float16)Wm2[K * 16 + col];
  }
}

// xwf[k][mc][nt][lane][h] = sum_i x[m][i]*aggW[k][i][o]  (B fragments, f16)
__global__ void xw_frag_kernel(const float* __restrict__ x, const float* __restrict__ aggW,
                               _Float16* __restrict__ xwf) {
  int t = blockIdx.x * blockDim.x + threadIdx.x;      // 524288
  int h = t & 15, lane = (t >> 4) & 31, nt = (t >> 9) & 1, mc = (t >> 10) & 31, k = t >> 15;
  int m = mc * 32 + h + ((lane < 16) ? 0 : 16);
  int o = nt * 16 + (lane & 15);
  float s = 0.f;
  #pragma unroll
  for (int i = 0; i < NN; ++i) s += x[m * NN + i] * aggW[(k * NN + i) * NNOUT + o];
  xwf[t] = (_Float16)s;
}

// per-edge linears + tmp assembly + scatter Xe/Ye into dense f16 Ax / AyT
__global__ void edge_pre_kernel(const float* __restrict__ C, const int* __restrict__ ei,
                                const float* __restrict__ W1, const float* __restrict__ W2,
                                const float* __restrict__ W3, const float* __restrict__ W4,
                                const float* __restrict__ h5,
                                _Float16* __restrict__ Ax, _Float16* __restrict__ AyT,
                                _Float16* __restrict__ tmp) {
  int t = blockIdx.x * blockDim.x + threadIdx.x;      // 524288
  int e = t >> 4, kcol = t & 15;
  int i = ei[e], j = ei[N_EDGES + e];
  const float* crow = C + (size_t)e * NE;
  float l1 = 0.f, l2 = 0.f, l3 = 0.f, l4 = 0.f;
  #pragma unroll
  for (int r = 0; r < NE; ++r) {
    float cr = crow[r];
    l1 += cr * W1[r * 16 + kcol];
    l2 += cr * W2[r * 16 + kcol];
    l3 += cr * W3[r * 16 + kcol];
    l4 += cr * W4[r * 16 + kcol];
  }
  const float inv = 1.0f / NE;
  float had = (l1 * inv) * (l2 * inv);
  float Xe = l3 * inv, Ye = l4 * inv;
  _Float16* trow = tmp + (size_t)e * TMPW;
  trow[kcol]      = (_Float16)crow[kcol];
  trow[16 + kcol] = (_Float16)had;
  bool diag = (i == j);
  trow[32 + kcol] = (_Float16)(diag ? h5[i * 32 + kcol]      : 0.f);
  trow[48 + kcol] = (_Float16)(diag ? h5[i * 32 + kcol + 16] : 0.f);
  trow[80 + kcol] = (_Float16)0.f;    // pad (col 64..79 written by matmul kernel)
  size_t ks = (size_t)kcol << 20;
  Ax [ks + (size_t)i * N_NODES + j] = (_Float16)Xe;
  AyT[ks + (size_t)j * N_NODES + i] = (_Float16)Ye;
}

// tmp_matmul[e,k] = dot(Ax[k][i_e,:], AyT[k][j_e,:]) via 16-edge-tile WMMA, keep diagonal
__global__ void __launch_bounds__(128)
edge_matmul_wmma(const _Float16* __restrict__ Ax, const _Float16* __restrict__ AyT,
                 const int* __restrict__ ei, _Float16* __restrict__ tmp) {
  int lane = threadIdx.x & 31;
  int task = blockIdx.x * 4 + (threadIdx.x >> 5);     // 32768 = 2048 tiles * 16 k
  int tile = task >> 4, k = task & 15;
  int eM = tile * 16 + (lane & 15);
  const _Float16* arow = Ax  + ((size_t)k << 20) + (size_t)ei[eM]           * N_NODES;
  const _Float16* brow = AyT + ((size_t)k << 20) + (size_t)ei[N_EDGES + eM] * N_NODES;
  int bhs = (lane >= 16) ? 16 : 0;
  v8f c = {};
  for (int m = 0; m < N_NODES; m += 32) {
    if (m + 128 < N_NODES) {
      __builtin_prefetch(arow + m + 128, 0, 1);       // global_prefetch_b8
      __builtin_prefetch(brow + m + 128, 0, 1);
    }
    v16h a = load_a_frag(arow, lane, m);
    v16h b = load_b_contig(brow + m + bhs);
    c = wmma_f16(a, b, c);
  }
  // diagonal of D: (e,e): e<8 -> lane=e, vgpr=e ; e>=8 -> lane=e+16, vgpr=e-8
  int idx = (lane < 8) ? lane : (lane - 24);
  float d = 0.f;
  #pragma unroll
  for (int r = 0; r < 8; ++r) d = (idx == r) ? c[r] : d;
  if (lane < 8 || lane >= 24) {
    int e = (lane < 8) ? lane : (lane - 16);
    tmp[(size_t)(tile * 16 + e) * TMPW + 64 + k] = (_Float16)d;
  }
}

// Cout = relu(tmp @ Wm1) @ Wm2 ; one wave per 16-edge tile; hidden kept in LDS
__global__ void __launch_bounds__(128)
edge_mlp_wmma(const _Float16* __restrict__ tmp, const _Float16* __restrict__ wm1f,
              const _Float16* __restrict__ wm2f, float* __restrict__ coutg) {
  __shared__ _Float16 hid[4][16 * 128];
  int lane = threadIdx.x & 31, wv = threadIdx.x >> 5;
  int tile = blockIdx.x * 4 + wv;                     // 2048 tiles
  int laneN = lane & 15;
  int mhalf = (lane < 16) ? 0 : 8;
  const _Float16* trow = tmp + (size_t)(tile * 16 + laneN) * TMPW;
  #pragma unroll
  for (int nt = 0; nt < 8; ++nt) {
    v8f c = {};
    #pragma unroll
    for (int kc = 0; kc < 3; ++kc) {
      v16h a = load_a_frag(trow, lane, kc * 32);
      v16h b = load_b_contig(wm1f + ((size_t)((nt * 3 + kc) * 32 + lane) << 4));
      c = wmma_f16(a, b, c);
    }
    #pragma unroll
    for (int r = 0; r < 8; ++r) {
      float v = c[r];
      hid[wv][(r + mhalf) * 128 + nt * 16 + laneN] = (_Float16)(v > 0.f ? v : 0.f);
    }
  }
  __syncthreads();
  v8f c2 = {};
  const _Float16* hrow = &hid[wv][laneN * 128];
  #pragma unroll
  for (int kc = 0; kc < 4; ++kc) {
    v16h a = load_a_frag(hrow, lane, kc * 32);
    v16h b = load_b_contig(wm2f + ((size_t)(kc * 32 + lane) << 4));
    c2 = wmma_f16(a, b, c2);
  }
  #pragma unroll
  for (int r = 0; r < 8; ++r)
    coutg[(size_t)(tile * 16 + r + mhalf) * NEOUT + laneN] = c2[r];
}

// scatter Cout into reused dense buffer for the aggregation GEMM
__global__ void scatter_cout_kernel(const float* __restrict__ coutg, const int* __restrict__ ei,
                                    _Float16* __restrict__ Adense) {
  int t = blockIdx.x * blockDim.x + threadIdx.x;      // 524288
  int e = t >> 4, k = t & 15;
  int i = ei[e], j = ei[N_EDGES + e];
  Adense[((size_t)k << 20) + (size_t)i * N_NODES + j] = (_Float16)coutg[(size_t)e * NEOUT + k];
}

// partial[k] = Adense[k] @ xw[k]  (16 batched 1024x1024x32 WMMA GEMMs, deterministic)
__global__ void __launch_bounds__(128)
agg_wmma(const _Float16* __restrict__ Adense, const _Float16* __restrict__ xwf,
         float* __restrict__ partial) {
  int lane = threadIdx.x & 31;
  int task = blockIdx.x * 4 + (threadIdx.x >> 5);     // 2048 = 64 rb * 2 nt * 16 k
  int k = task & 15, nt = (task >> 4) & 1, rb = task >> 5;
  const _Float16* arow = Adense + ((size_t)k << 20) + (size_t)(rb * 16 + (lane & 15)) * N_NODES;
  v8f c = {};
  for (int mc = 0; mc < 32; ++mc) {
    if (mc + 4 < 32) __builtin_prefetch(arow + (mc + 4) * 32, 0, 1);
    v16h a = load_a_frag(arow, lane, mc * 32);
    v16h b = load_b_contig(xwf + ((size_t)(((k * 32 + mc) * 2 + nt) * 32 + lane) << 4));
    c = wmma_f16(a, b, c);
  }
  int mhalf = (lane < 16) ? 0 : 8;
  #pragma unroll
  for (int r = 0; r < 8; ++r) {
    int n = rb * 16 + r + mhalf;
    int o = nt * 16 + (lane & 15);
    partial[(size_t)k * (N_NODES * NNOUT) + n * NNOUT + o] = c[r];
  }
}

// xout = (sum_k partial[k] + bias) / 32
__global__ void finalize_kernel(const float* __restrict__ partial, const float* __restrict__ bias,
                                float* __restrict__ xout) {
  int t = blockIdx.x * blockDim.x + threadIdx.x;      // 32768
  float s = bias[t & 31];
  #pragma unroll
  for (int k = 0; k < NEOUT; ++k) s += partial[(size_t)k * (N_NODES * NNOUT) + t];
  xout[t] = s * (1.0f / NN);
}

// ---------------------------------------------------------------------------
extern "C" void kernel_launch(void* const* d_in, const int* in_sizes, int n_in,
                              void* d_out, int out_size, void* d_ws, size_t ws_size,
                              hipStream_t stream) {
  (void)in_sizes; (void)n_in; (void)out_size; (void)ws_size;
  const float* x    = (const float*)d_in[0];
  const float* C    = (const float*)d_in[1];
  const int*   ei   = (const int*)d_in[3];
  const float* W1   = (const float*)d_in[4];
  const float* W2   = (const float*)d_in[5];
  const float* W3   = (const float*)d_in[6];
  const float* W4   = (const float*)d_in[7];
  const float* W5   = (const float*)d_in[8];
  const float* Wm1  = (const float*)d_in[9];
  const float* Wm2  = (const float*)d_in[10];
  const float* aggW = (const float*)d_in[11];
  const float* aggB = (const float*)d_in[12];

  char* ws = (char*)d_ws;
  _Float16* Ax   = (_Float16*)(ws + AX_OFF);
  _Float16* AyT  = (_Float16*)(ws + AYT_OFF);
  _Float16* tmp  = (_Float16*)(ws + TMP_OFF);
  float*    h5   = (float*)   (ws + H5_OFF);
  _Float16* wm1f = (_Float16*)(ws + WM1F_OFF);
  _Float16* wm2f = (_Float16*)(ws + WM2F_OFF);
  _Float16* xwf  = (_Float16*)(ws + XWF_OFF);
  float*    part = (float*)   (ws + PART_OFF);

  float* xout  = (float*)d_out;                  // [1024][32]
  float* coutg = (float*)d_out + N_NODES * NNOUT; // [E][16]

  // 1) zero dense Ax + AyT (contiguous 64MB)
  zero_u4_kernel<<<16384, 256, 0, stream>>>((uint4*)Ax, 4194304);
  // 2) small precomputes
  h5_kernel<<<128, 256, 0, stream>>>(x, W5, h5);
  weight_frag_kernel<<<56, 256, 0, stream>>>(Wm1, Wm2, wm1f, wm2f);
  xw_frag_kernel<<<2048, 256, 0, stream>>>(x, aggW, xwf);
  // 3) per-edge linears + scatter + tmp assembly
  edge_pre_kernel<<<2048, 256, 0, stream>>>(C, ei, W1, W2, W3, W4, h5, Ax, AyT, tmp);
  // 4) sparse-gathered batched GEMM diagonal (WMMA)
  edge_matmul_wmma<<<8192, 128, 0, stream>>>(Ax, AyT, ei, tmp);
  // 5) edge MLP (WMMA) -> Cout directly into d_out
  edge_mlp_wmma<<<512, 128, 0, stream>>>(tmp, wm1f, wm2f, coutg);
  // 6) aggregation: zero + scatter Cout + 16 batched GEMMs (WMMA) + reduce
  zero_u4_kernel<<<8192, 256, 0, stream>>>((uint4*)Ax, 2097152);
  scatter_cout_kernel<<<2048, 256, 0, stream>>>(coutg, ei, Ax);
  agg_wmma<<<512, 128, 0, stream>>>(Ax, xwf, part);
  finalize_kernel<<<128, 256, 0, stream>>>(part, aggB, xout);
}
